// ThresholdedAttention_34282428957119
// MI455X (gfx1250) — matrix operand order repaired and checked by reading, not compiled, not run on hardware
//
#include <hip/hip_runtime.h>

// ---- CDNA5 (gfx1250) wave32 WMMA attention implementation ----
// Pipeline: f32->f16 convert, QKV GEMMs (WMMA + TDM-staged weight tiles),
// flash attention (WMMA + TDM-staged K tiles), output GEMM.
// Matmuls: v_wmma_f32_16x16x32_f16. Tile staging: tensor_load_to_lds (TDM,
// TENSORcnt) where available, cooperative loads otherwise.

typedef __attribute__((ext_vector_type(16))) _Float16     v16h;
typedef __attribute__((ext_vector_type(8)))  _Float16     v8h;
typedef __attribute__((ext_vector_type(8)))  float        v8f;
typedef __attribute__((ext_vector_type(4)))  unsigned int v4u;
typedef __attribute__((ext_vector_type(8)))  int          v8i;
typedef __attribute__((ext_vector_type(4)))  int          v4i;

#define WMMA_F16(a, b, c) \
  __builtin_amdgcn_wmma_f32_16x16x32_f16(false, (a), false, (b), (short)0, (c), false, false)

#if __has_builtin(__builtin_amdgcn_tensor_load_to_lds) && __has_builtin(__builtin_amdgcn_s_wait_tensorcnt)
#define HAVE_TDM 1
#else
#define HAVE_TDM 0
#endif

#if HAVE_TDM
// Issue a 2D TDM tile load: tile_d1 rows x tile_d0 elements (f16, 2 bytes)
// from a row-major tensor with row stride stride_d0 (elements), into LDS at
// byte offset lds_addr (rows stored consecutively, tile_d0*2 bytes apart).
// D# layout per CDNA5 ISA ch.8 (group0: count/type/addresses, group1: dims).
// This toolchain exposes the 6-arg builtin:
//   (uint32x4 g0, int32x8 g1, int32x4 g2, int32x4 g3, int32x8 extra, i32 cpol)
__device__ __forceinline__ void tdm_load_2d(unsigned lds_addr, const void* gaddr,
                                            unsigned tensor_d0, unsigned tensor_d1,
                                            unsigned tile_d0, unsigned tile_d1,
                                            unsigned stride_d0) {
  unsigned long long ga = (unsigned long long)gaddr;
  v4u g0;
  g0[0] = 1u;                                           // count=1 (valid), user mode
  g0[1] = lds_addr;                                     // lds_addr [63:32]
  g0[2] = (unsigned)(ga & 0xFFFFFFFFu);                 // global_addr [95:64]
  g0[3] = (unsigned)((ga >> 32) & 0x1FFFFFFu)           // global_addr [120:96]
        | (2u << 30);                                   // type=2 ("image") [127:126]
  v8i g1;
  g1[0] = (int)(1u << 16);                              // data_size=1 -> 2 bytes
  g1[1] = (int)((tensor_d0 & 0xFFFFu) << 16);           // tensor_dim0[15:0] @ bit48
  g1[2] = (int)((tensor_d0 >> 16) | ((tensor_d1 & 0xFFFFu) << 16)); // d0 hi | d1 lo
  g1[3] = (int)((tensor_d1 >> 16) | (tile_d0 << 16));   // d1 hi | tile_dim0
  g1[4] = (int)(tile_d1 & 0xFFFFu);                     // tile_dim1 | tile_dim2=0
  g1[5] = (int)stride_d0;                               // tensor_dim0_stride[31:0]
  g1[6] = 0;                                            // stride hi | d1 stride lo
  g1[7] = 0;
  v4i z4 = {0, 0, 0, 0};
  v8i z8 = {0, 0, 0, 0, 0, 0, 0, 0};
  __builtin_amdgcn_tensor_load_to_lds(g0, g1, z4, z4, z8, 0);
}
#endif

__device__ __forceinline__ unsigned lds_offset_of(const void* p) {
  return (unsigned)(unsigned long long)p;   // low 32 bits of flat addr = LDS offset
}

// --------------------------------------------------------------------------
// f32 -> f16 conversion
// --------------------------------------------------------------------------
__global__ void __launch_bounds__(256)
cvt_f32_to_f16_kernel(const float* __restrict__ in, _Float16* __restrict__ out, int n) {
  int i = (blockIdx.x * 256 + threadIdx.x) * 4;
  if (i + 3 < n) {
    float4 f = *(const float4*)(in + i);
    out[i + 0] = (_Float16)f.x;
    out[i + 1] = (_Float16)f.y;
    out[i + 2] = (_Float16)f.z;
    out[i + 3] = (_Float16)f.w;
  } else {
    for (; i < n; ++i) out[i] = (_Float16)in[i];
  }
}

// --------------------------------------------------------------------------
// GEMM: C[M,N] = A[M,K] * W[N,K]^T + bias
// Block: 256 threads = 8 waves; wave computes 16x64 strip; block tile 128x64.
// Weight k-tile (64 rows x 128 k, 16KB) staged in LDS via TDM.
// mode 0: f32 row-major [M,N]; mode 1: f16 [B,H,T,D] (T=2048,H=16,D=64).
// --------------------------------------------------------------------------
__global__ void __launch_bounds__(256)
gemm_f16_kernel(const _Float16* __restrict__ A,   // [M,K]
                const _Float16* __restrict__ W,   // [N,K] row-major
                const float*    __restrict__ bias,
                float*          __restrict__ outF,
                _Float16*       __restrict__ outH,
                int M, int N, int K, int mode) {
  __attribute__((aligned(128))) __shared__ _Float16 Ws[64 * 128];  // [n_local][k_local]

  const int lane = threadIdx.x & 31;
  const int wave = threadIdx.x >> 5;
  const int half = lane >> 4;   // 0: lanes 0-15, 1: lanes 16-31
  const int l16  = lane & 15;

  const int rowBase = blockIdx.x * 128 + wave * 16;
  const int colBase = blockIdx.y * 64;

  v8f c[4] = {};
  const _Float16* Arow = A + (size_t)(rowBase + l16) * K;

  for (int kt = 0; kt < K; kt += 128) {
    __syncthreads();   // previous tile fully consumed
#if HAVE_TDM
    if (wave == 0) {
      tdm_load_2d(lds_offset_of(&Ws[0]),
                  W + (size_t)colBase * K + kt,
                  (unsigned)K, (unsigned)N,   // tensor dims (elements)
                  128u, 64u,                  // tile: 128 k-elems x 64 rows
                  (unsigned)K);               // row stride
      __builtin_amdgcn_s_wait_tensorcnt(0);
    }
#else
    {
      // cooperative fallback: 64 rows x 8 chunks of 16 f16
      for (int i = threadIdx.x; i < 64 * 8; i += 256) {
        const int r  = i >> 3;
        const int cb = (i & 7) * 16;
        *(v16h*)(&Ws[r * 128 + cb]) =
            *(const v16h*)(W + (size_t)(colBase + r) * K + kt + cb);
      }
    }
#endif
    // prefetch next A stream chunk while the tile stages
    if (kt + 128 < K) __builtin_prefetch((const void*)(Arow + kt + 128), 0, 0);
    __syncthreads();

#pragma unroll
    for (int kc = 0; kc < 4; ++kc) {
      const int k0 = kt + kc * 32;
      v16h a;
      {
        union { v16h v; v8h h[2]; } u;
        u.h[0] = *(const v8h*)(Arow + k0 + half * 16);
        u.h[1] = *(const v8h*)(Arow + k0 + half * 16 + 8);
        a = u.v;
      }
#pragma unroll
      for (int s = 0; s < 4; ++s) {
        v16h b = *(const v16h*)(&Ws[(s * 16 + l16) * 128 + kc * 32 + half * 16]);
        c[s] = WMMA_F16(a, b, c[s]);
      }
    }
  }

  // Epilogue. C layout: VGPR r -> row (r + 8*half); lane -> col l16.
#pragma unroll
  for (int s = 0; s < 4; ++s) {
    const int col = colBase + s * 16 + l16;
    const float bv = bias[col];
#pragma unroll
    for (int r = 0; r < 8; ++r) {
      const int row = rowBase + r + half * 8;
      const float v = c[s][r] + bv;
      if (mode == 0) {
        outF[(size_t)row * N + col] = v;
      } else {
        const int b_ = row >> 11;   // / 2048
        const int t  = row & 2047;
        const int h  = col >> 6;    // / 64
        const int d  = col & 63;
        outH[(((size_t)b_ * 16 + h) * 2048 + t) * 64 + d] = (_Float16)v;
      }
    }
  }
}

// --------------------------------------------------------------------------
// Flash attention with causal mask + per-head score threshold.
// Grid: (T/128, B*H). Block: 256 threads = 8 waves; wave w handles q rows
// [blockIdx.x*128 + w*16, +16). kv tiles of 32 columns. K tile via TDM.
// --------------------------------------------------------------------------
__global__ void __launch_bounds__(256)
flash_attn_kernel(const _Float16* __restrict__ Q,   // [B,H,T,D]
                  const _Float16* __restrict__ Km,  // [B,H,T,D]
                  const _Float16* __restrict__ Vm,  // [B,H,T,D]
                  _Float16*       __restrict__ Oh,  // [B,T,C] f16
                  const float*    __restrict__ attn_scale,
                  const float*    __restrict__ score_threshold,
                  const float*    __restrict__ running_var) {
  const int T = 2048, D = 64, C = 1024;
  __attribute__((aligned(128))) __shared__ _Float16 Kt[32 * 64];      // [kv][d]
  __attribute__((aligned(128))) __shared__ _Float16 Vt[64 * 32];      // transposed [d][kv]
  __attribute__((aligned(128))) __shared__ _Float16 Pt[8][16 * 32];   // per-wave P staging

  const int lane = threadIdx.x & 31;
  const int wave = threadIdx.x >> 5;
  const int half = lane >> 4;
  const int l16  = lane & 15;

  const int bh = blockIdx.y;        // b*H + h
  const int b  = bh >> 4;
  const int h  = bh & 15;
  const int qb = blockIdx.x * 128 + wave * 16;

  const float scale = attn_scale[0];
  const float st    = score_threshold[0];
  const float thr   = (1.0f / (1.0f + __expf(-st))) * sqrtf(fmaxf(running_var[h], 1e-6f));

  const _Float16* Qbase = Q  + (size_t)bh * T * D;
  const _Float16* Kbase = Km + (size_t)bh * T * D;
  const _Float16* Vbase = Vm + (size_t)bh * T * D;

  // Q fragments for D=64 (two K-chunks of 32), resident for the whole loop.
  v16h qa[2];
  {
    const _Float16* qrow = Qbase + (size_t)(qb + l16) * D;
#pragma unroll
    for (int ck = 0; ck < 2; ++ck) {
      union { v16h v; v8h hh[2]; } u;
      u.hh[0] = *(const v8h*)(qrow + ck * 32 + half * 16);
      u.hh[1] = *(const v8h*)(qrow + ck * 32 + half * 16 + 8);
      qa[ck] = u.v;
    }
  }

  v8f o[4] = {};
  float mrow[8], lrow[8];
#pragma unroll
  for (int r = 0; r < 8; ++r) { mrow[r] = -__builtin_inff(); lrow[r] = 0.0f; }

  const int ntiles = (blockIdx.x * 128 + 128) / 32;  // uniform per block
  for (int tile = 0; tile < ntiles; ++tile) {
    const int kv0 = tile * 32;

#if HAVE_TDM
    if (wave == 0) {
      tdm_load_2d(lds_offset_of(&Kt[0]),
                  Kbase + (size_t)kv0 * D,
                  (unsigned)D, (unsigned)T,   // tensor dims
                  (unsigned)D, 32u,           // tile: 64 d-elems x 32 kv rows
                  (unsigned)D);
    }
#else
    {
      const int row = threadIdx.x >> 3;
      const int cb  = (threadIdx.x & 7) * 8;
      *(v8h*)(&Kt[row * 64 + cb]) =
          *(const v8h*)(Kbase + (size_t)(kv0 + row) * D + cb);
    }
#endif
    // V tile load + transpose (overlaps the TDM K-tile DMA)
    {
      const int row = threadIdx.x >> 3;        // 0..31 kv index within tile
      const int cb  = (threadIdx.x & 7) * 8;   // d block
      v8h vv = *(const v8h*)(Vbase + (size_t)(kv0 + row) * D + cb);
#pragma unroll
      for (int j = 0; j < 8; ++j) Vt[(cb + j) * 32 + row] = vv[j];
    }
#if HAVE_TDM
    if (wave == 0) __builtin_amdgcn_s_wait_tensorcnt(0);
#endif
    __syncthreads();

    // S = Q * K^T for this wave's 16 q rows; two 16-col fragments.
    v8f s0 = {}, s1 = {};
#pragma unroll
    for (int ck = 0; ck < 2; ++ck) {
      v16h b0 = *(const v16h*)(&Kt[l16 * 64        + ck * 32 + half * 16]);
      v16h b1 = *(const v16h*)(&Kt[(16 + l16) * 64 + ck * 32 + half * 16]);
      s0 = WMMA_F16(qa[ck], b0, s0);
      s1 = WMMA_F16(qa[ck], b1, s1);
    }

    // scale + causal mask + per-head threshold
    float p0[8], p1[8], tmax[8];
#pragma unroll
    for (int r = 0; r < 8; ++r) {
      const int qi  = qb + r + half * 8;
      const int k0i = kv0 + l16;
      const int k1i = kv0 + 16 + l16;
      float a0 = s0[r] * scale;
      float a1 = s1[r] * scale;
      if (k0i > qi) a0 = -__builtin_inff(); else if (a0 < thr) a0 = -10000.0f;
      if (k1i > qi) a1 = -__builtin_inff(); else if (a1 < thr) a1 = -10000.0f;
      p0[r] = a0; p1[r] = a1;
      tmax[r] = fmaxf(a0, a1);
    }
    // row max across the 16 lanes sharing each row
#pragma unroll
    for (int r = 0; r < 8; ++r) {
      float v = tmax[r];
      v = fmaxf(v, __shfl_xor(v, 1, 32));
      v = fmaxf(v, __shfl_xor(v, 2, 32));
      v = fmaxf(v, __shfl_xor(v, 4, 32));
      v = fmaxf(v, __shfl_xor(v, 8, 32));
      tmax[r] = v;
    }
    // online softmax update
    float alpha[8];
#pragma unroll
    for (int r = 0; r < 8; ++r) {
      const float mnew = fmaxf(mrow[r], tmax[r]);
      float al, e0, e1;
      if (mnew == -__builtin_inff()) { al = 1.0f; e0 = 0.0f; e1 = 0.0f; }
      else {
        al = __expf(mrow[r] - mnew);   // exp(-inf)=0 handles first tile
        e0 = __expf(p0[r] - mnew);
        e1 = __expf(p1[r] - mnew);
      }
      float rs = e0 + e1;
      rs += __shfl_xor(rs, 1, 32);
      rs += __shfl_xor(rs, 2, 32);
      rs += __shfl_xor(rs, 4, 32);
      rs += __shfl_xor(rs, 8, 32);
      lrow[r] = al * lrow[r] + rs;
      mrow[r] = mnew;
      alpha[r] = al;
      p0[r] = e0; p1[r] = e1;
    }
    // rescale O accumulators
#pragma unroll
    for (int s = 0; s < 4; ++s)
#pragma unroll
      for (int r = 0; r < 8; ++r) o[s][r] *= alpha[r];

    // stage P (C-layout -> [m][k]) in LDS, reload as A-fragment
    _Float16* P = &Pt[wave][0];
#pragma unroll
    for (int r = 0; r < 8; ++r) {
      const int m = r + half * 8;
      P[m * 32 + l16]      = (_Float16)p0[r];
      P[m * 32 + 16 + l16] = (_Float16)p1[r];
    }
    __syncthreads();

    v16h pa;
    {
      union { v16h v; v8h hh[2]; } u;
      u.hh[0] = *(const v8h*)(&P[l16 * 32 + half * 16]);
      u.hh[1] = *(const v8h*)(&P[l16 * 32 + half * 16 + 8]);
      pa = u.v;
    }
#pragma unroll
    for (int s = 0; s < 4; ++s) {
      v16h vb = *(const v16h*)(&Vt[(s * 16 + l16) * 32 + half * 16]);
      o[s] = WMMA_F16(pa, vb, o[s]);
    }
    __syncthreads();  // protect Kt/Vt/Pt before next tile's writes
  }

  // normalize and write [B,T,C] f16
#pragma unroll
  for (int s = 0; s < 4; ++s) {
#pragma unroll
    for (int r = 0; r < 8; ++r) {
      const int t = qb + r + half * 8;
      const int d = s * 16 + l16;
      const float v = o[s][r] / lrow[r];
      Oh[((size_t)b * T + t) * C + h * 64 + d] = (_Float16)v;
    }
  }
}

// --------------------------------------------------------------------------
// Launch
// --------------------------------------------------------------------------
extern "C" void kernel_launch(void* const* d_in, const int* in_sizes, int n_in,
                              void* d_out, int out_size, void* d_ws, size_t ws_size,
                              hipStream_t stream) {
  (void)in_sizes; (void)n_in; (void)out_size; (void)ws_size;
  const int B = 2, T = 2048, C = 1024;
  const int M = B * T;  // 4096

  const float* x   = (const float*)d_in[0];
  // d_in[1] = attn_mask (causal; computed analytically in-kernel)
  const float* Wq  = (const float*)d_in[2];
  const float* bq  = (const float*)d_in[3];
  const float* Wk  = (const float*)d_in[4];
  const float* bk  = (const float*)d_in[5];
  const float* Wv  = (const float*)d_in[6];
  const float* bv  = (const float*)d_in[7];
  const float* Wo  = (const float*)d_in[8];
  const float* bo  = (const float*)d_in[9];
  const float* asc = (const float*)d_in[10];
  const float* sth = (const float*)d_in[11];
  const float* rv  = (const float*)d_in[12];

  // Workspace carve-up (all f16 = 2 bytes/elem)
  char* ws = (char*)d_ws;
  const size_t xcElems = (size_t)M * C;          // 4,194,304
  const size_t wElems  = (size_t)C * C;          // 1,048,576
  _Float16* xh  = (_Float16*)(ws);                                       // M*C
  _Float16* Wqh = (_Float16*)(ws + xcElems * 2);                         // C*C
  _Float16* Wkh = (_Float16*)(ws + xcElems * 2 + wElems * 2);
  _Float16* Wvh = (_Float16*)(ws + xcElems * 2 + wElems * 4);
  _Float16* Woh = (_Float16*)(ws + xcElems * 2 + wElems * 6);
  _Float16* Qh  = (_Float16*)(ws + xcElems * 2 + wElems * 8);            // [B,H,T,D]
  _Float16* Kh  = (_Float16*)(ws + xcElems * 4 + wElems * 8);
  _Float16* Vh  = (_Float16*)(ws + xcElems * 6 + wElems * 8);
  _Float16* Ah  = (_Float16*)(ws + xcElems * 8 + wElems * 8);            // attn out [B,T,C]

  // f32 -> f16 conversions
  {
    const int nx = (int)xcElems;
    cvt_f32_to_f16_kernel<<<(nx / 4 + 255) / 256, 256, 0, stream>>>(x, xh, nx);
    const int nw = (int)wElems;
    cvt_f32_to_f16_kernel<<<(nw / 4 + 255) / 256, 256, 0, stream>>>(Wq, Wqh, nw);
    cvt_f32_to_f16_kernel<<<(nw / 4 + 255) / 256, 256, 0, stream>>>(Wk, Wkh, nw);
    cvt_f32_to_f16_kernel<<<(nw / 4 + 255) / 256, 256, 0, stream>>>(Wv, Wvh, nw);
    cvt_f32_to_f16_kernel<<<(nw / 4 + 255) / 256, 256, 0, stream>>>(Wo, Woh, nw);
  }

  // QKV projections (mode 1 -> [B,H,T,D] f16)
  dim3 gg(M / 128, C / 64);
  gemm_f16_kernel<<<gg, 256, 0, stream>>>(xh, Wqh, bq, nullptr, Qh, M, C, C, 1);
  gemm_f16_kernel<<<gg, 256, 0, stream>>>(xh, Wkh, bk, nullptr, Kh, M, C, C, 1);
  gemm_f16_kernel<<<gg, 256, 0, stream>>>(xh, Wvh, bv, nullptr, Vh, M, C, C, 1);

  // attention
  dim3 ga(T / 128, B * 16);
  flash_attn_kernel<<<ga, 256, 0, stream>>>(Qh, Kh, Vh, Ah, asc, sth, rv);

  // output projection (mode 0 -> f32 [B,T,C] = d_out)
  gemm_f16_kernel<<<gg, 256, 0, stream>>>(Ah, Woh, bo, (float*)d_out, nullptr, M, C, C, 0);
}
